// Mymodel_91293824844107
// MI455X (gfx1250) — matrix operand order repaired and checked by reading, not compiled, not run on hardware
//
#include <hip/hip_runtime.h>
#include <hip/hip_bf16.h>
#include <math.h>

// ---------------------------------------------------------------------------
// Types for CDNA5 WMMA (gfx1250, wave32)
// ---------------------------------------------------------------------------
typedef __attribute__((ext_vector_type(16))) __bf16 v16bf;
typedef __attribute__((ext_vector_type(8)))  __bf16 v8bf;
typedef __attribute__((ext_vector_type(8)))  float  v8f;
typedef __attribute__((ext_vector_type(8)))  float  v8fl;

#define HID        128
#define G4         512          // 4 * HID
#define PATHS      4096
#define PLEN       7
#define WAVES_WG   8            // 256 threads
#define PATHS_WG   128          // 16 paths per wave
#define WGS_PER_MP 32           // 4096 / 128
#define NUM_WG     (4 * WGS_PER_MP)
#define HROW       136          // padded LDS h row stride (bf16 elems)
#define CROW       132          // padded LDS c row stride (f32 elems)

__device__ __forceinline__ float sigm(float x) { return 1.0f / (1.0f + __expf(-x)); }

// ---------------------------------------------------------------------------
// Kernel 1: convert LSTM weights f32 -> bf16 (row-major [512][128]).
// ---------------------------------------------------------------------------
__global__ __launch_bounds__(256) void prep_weights(
    const float* __restrict__ wih_f, const float* __restrict__ whh_f,
    __bf16* __restrict__ wih, __bf16* __restrict__ whh) {
  int i = blockIdx.x * 256 + threadIdx.x;
  if (i < G4 * HID) {
    wih[i] = (__bf16)wih_f[i];
    whh[i] = (__bf16)whh_f[i];
  }
}

// ---------------------------------------------------------------------------
// Kernel 2: fused gather + 7-step LSTM + per-WG path-sum of final h.
// grid = 128 WGs (32 per metapath), block = 256 threads = 8 waves.
// hj loop is deliberately rolled so the 256 loop-invariant weight B-fragments
// cannot be hoisted+spilled to scratch; c state lives in LDS instead of a
// dynamically-indexed private array.
// ---------------------------------------------------------------------------
__global__ __launch_bounds__(256) void metapath_lstm(
    const int* __restrict__ ids0, const int* __restrict__ ids1,
    const int* __restrict__ ids2, const int* __restrict__ ids3,
    const float* __restrict__ emb,
    const __bf16* __restrict__ wih, const __bf16* __restrict__ whh,
    const float* __restrict__ b_ih, const float* __restrict__ b_hh,
    float* __restrict__ partials) {

  __shared__ alignas(16) __bf16 hbuf[WAVES_WG * 16 * HROW]; // wave-private h (bf16)
  __shared__ alignas(16) float  cbuf[WAVES_WG * 16 * CROW]; // wave-private c (f32)
  __shared__ float biasLds[G4];
  __shared__ float red[WAVES_WG][HID];

  const int wg   = blockIdx.x;
  const int mp   = wg >> 5;                 // metapath 0..3
  const int tile = wg & 31;                 // path tile within metapath
  const int wave = threadIdx.x >> 5;
  const int lane = threadIdx.x & 31;
  const int lrow = lane & 15;               // M row (A) or N col (B/C) for this lane
  const bool lo  = lane < 16;
  const int pathBase = tile * PATHS_WG + wave * 16;

  const int* ids = (mp == 0) ? ids0 : (mp == 1) ? ids1 : (mp == 2) ? ids2 : ids3;

  // combined bias (b_ih + b_hh), cooperative
  for (int j = threadIdx.x; j < G4; j += 256)
    biasLds[j] = b_ih[j] + b_hh[j];

  // zero wave-private h / c state (h0 = c0 = 0)
  __bf16* myh = hbuf + wave * 16 * HROW;
  float*  myc = cbuf + wave * 16 * CROW;
  for (int j = lane; j < 16 * HROW; j += 32) myh[j] = (__bf16)0.0f;
  for (int j = lane; j < 16 * CROW; j += 32) myc[j] = 0.0f;
  __syncthreads();

#pragma unroll 1
  for (int t = 0; t < PLEN; ++t) {
    // ---- gather x row for this lane's path directly into A-fragments ----
    const int node = ids[(pathBase + lrow) * PLEN + t];
    const float* xrow = emb + (size_t)node * HID;

    // prefetch next timestep's embedding row (covers the 512B row)
    if (t + 1 < PLEN) {
      const int nn = ids[(pathBase + lrow) * PLEN + t + 1];
      const char* nrow = (const char*)(emb + (size_t)nn * HID);
      __builtin_prefetch(nrow, 0, 0);
      __builtin_prefetch(nrow + 256, 0, 0);
    }

    v16bf xf[4], hf[4];
#pragma unroll
    for (int kt = 0; kt < 4; ++kt) {
      // A K order used here (contiguous): lo lanes K0..15, hi lanes K16..31 (matches B)
      const int e0 = kt * 32 + (lo ? 0 : 16);
      v8fl c0 = *(const v8fl*)(xrow + e0);
      v8fl c1 = *(const v8fl*)(xrow + e0 + 8);
      v16bf xv;
#pragma unroll
      for (int i = 0; i < 8; ++i) { xv[i] = (__bf16)c0[i]; xv[i + 8] = (__bf16)c1[i]; }
      xf[kt] = xv;

      const __bf16* hr = myh + lrow * HROW;
      v8bf h0 = *(const v8bf*)(hr + e0);       // same K layout, bf16 elems
      v8bf h1 = *(const v8bf*)(hr + e0 + 8);
      v16bf hv;
#pragma unroll
      for (int i = 0; i < 8; ++i) { hv[i] = h0[i]; hv[i + 8] = h1[i]; }
      hf[kt] = hv;
    }

    const int ke = lo ? 0 : 16;  // B K order used here (contiguous, as A): lo lanes K0..15, hi lanes K16..31

#pragma unroll 1                 // ROLLED: keeps weight loads un-hoistable (no spills)
    for (int hj = 0; hj < 8; ++hj) {
      const int hid = hj * 16 + lrow;          // this lane's hidden unit in the tile

      v8f acc[4];
#pragma unroll
      for (int g = 0; g < 4; ++g) {
        v8f a = {};
        const int nrow = g * HID + hid;        // gate-unit row of W (N = lrow)
        const __bf16* wi = wih + (size_t)nrow * HID;
        const __bf16* wh = whh + (size_t)nrow * HID;
#pragma unroll
        for (int kt = 0; kt < 4; ++kt) {
          v16bf bi = *(const v16bf*)(wi + kt * 32 + ke);
          a = __builtin_amdgcn_wmma_f32_16x16x32_bf16(false, xf[kt], false, bi,
                                                      (short)0, a, false, false);
        }
#pragma unroll
        for (int kt = 0; kt < 4; ++kt) {
          v16bf bh = *(const v16bf*)(wh + kt * 32 + ke);
          a = __builtin_amdgcn_wmma_f32_16x16x32_bf16(false, hf[kt], false, bh,
                                                      (short)0, a, false, false);
        }
        acc[g] = a;
      }

      const float bi_ = biasLds[0 * HID + hid];
      const float bf_ = biasLds[1 * HID + hid];
      const float bg_ = biasLds[2 * HID + hid];
      const float bo_ = biasLds[3 * HID + hid];

      float hs = 0.0f;                         // in-lane sum of h over 8 path rows
#pragma unroll
      for (int r = 0; r < 8; ++r) {
        const float gi = sigm(acc[0][r] + bi_);
        const float gf = sigm(acc[1][r] + bf_);
        const float gg = tanhf(acc[2][r] + bg_);
        const float go = sigm(acc[3][r] + bo_);
        const int row = lo ? r : (r + 8);      // C/D layout: VGPR r -> M=r / M=r+8
        float* cp = myc + row * CROW + hid;
        const float c = gf * (*cp) + gi * gg;
        *cp = c;
        const float h = go * tanhf(c);
        myh[row * HROW + hid] = (__bf16)h;
        hs += h;
      }

      if (t == PLEN - 1) {                     // final step: reduce over 16 paths
        const float s = hs + __shfl_xor(hs, 16, 32);
        if (lo) red[wave][hid] = s;
      }
    }
  }

  __syncthreads();
  if (threadIdx.x < HID) {
    float s = 0.0f;
#pragma unroll
    for (int w = 0; w < WAVES_WG; ++w) s += red[w][threadIdx.x];
    partials[wg * HID + threadIdx.x] = s;      // deterministic per-WG partial
  }
}

// ---------------------------------------------------------------------------
// Kernel 3: mean over paths, max over metapaths, linear + sigmoid -> out[0]
// ---------------------------------------------------------------------------
__global__ __launch_bounds__(128) void finalize(
    const float* __restrict__ partials, const float* __restrict__ wlin,
    const float* __restrict__ blin, float* __restrict__ out) {
  __shared__ float dotbuf[HID];
  const int j = threadIdx.x;
  float rep = -INFINITY;
  for (int mp = 0; mp < 4; ++mp) {
    float s = 0.0f;
    for (int w = 0; w < WGS_PER_MP; ++w)
      s += partials[(mp * WGS_PER_MP + w) * HID + j];
    s *= (1.0f / (float)PATHS);
    rep = fmaxf(rep, s);
  }
  dotbuf[j] = rep * wlin[j];
  __syncthreads();
  for (int off = 64; off > 0; off >>= 1) {
    if (j < off) dotbuf[j] += dotbuf[j + off];
    __syncthreads();
  }
  if (j == 0) out[0] = 1.0f / (1.0f + __expf(-(dotbuf[0] + blin[0])));
}

// ---------------------------------------------------------------------------
extern "C" void kernel_launch(void* const* d_in, const int* in_sizes, int n_in,
                              void* d_out, int out_size, void* d_ws, size_t ws_size,
                              hipStream_t stream) {
  const int*   ids0  = (const int*)d_in[0];
  const int*   ids1  = (const int*)d_in[1];
  const int*   ids2  = (const int*)d_in[2];
  const int*   ids3  = (const int*)d_in[3];
  const float* emb   = (const float*)d_in[4];
  const float* W_ih  = (const float*)d_in[5];
  const float* W_hh  = (const float*)d_in[6];
  const float* b_ih  = (const float*)d_in[7];
  const float* b_hh  = (const float*)d_in[8];
  const float* W_lin = (const float*)d_in[9];
  const float* b_lin = (const float*)d_in[10];

  char* ws = (char*)d_ws;
  __bf16* wih_bf   = (__bf16*)(ws);                       // 128 KB
  __bf16* whh_bf   = (__bf16*)(ws + G4 * HID * 2);        // 128 KB
  float*  partials = (float*)(ws + 2 * G4 * HID * 2);     // 64 KB

  prep_weights<<<(G4 * HID + 255) / 256, 256, 0, stream>>>(W_ih, W_hh, wih_bf, whh_bf);
  metapath_lstm<<<NUM_WG, 256, 0, stream>>>(ids0, ids1, ids2, ids3, emb,
                                            wih_bf, whh_bf, b_ih, b_hh, partials);
  finalize<<<1, 128, 0, stream>>>(partials, W_lin, b_lin, (float*)d_out);
}
